// DynamicConvolution_45938970198200
// MI455X (gfx1250) — matrix operand order repaired and checked by reading, not compiled, not run on hardware
//
#include <hip/hip_runtime.h>
#include <hip/hip_bf16.h>

// ---------------- problem constants (from reference setup_inputs) ----------
#define B_   8
#define NS   4096     // scene points
#define NT   512      // template points
#define C_   128      // feature channels (SC == TC == 128)
#define TP   3201     // TOTAL_PARAMS = 3*1024 + 32 + 32+32+32+1

// ---------------- WMMA vector types ---------------------------------------
typedef __attribute__((ext_vector_type(16))) _Float16 v16h;
typedef __attribute__((ext_vector_type(8)))  _Float16 v8h;
typedef __attribute__((ext_vector_type(8)))  float    v8f;
typedef __attribute__((ext_vector_type(4)))  int      v4i;

// CDNA5 async global->LDS path (ASYNCcnt-tracked), guarded so the file
// compiles on toolchains that don't declare the builtins.
#if defined(__has_builtin)
#if __has_builtin(__builtin_amdgcn_global_load_async_to_lds_b128) && \
    __has_builtin(__builtin_amdgcn_s_wait_asynccnt)
#define USE_ASYNC_LDS 1
#endif
#endif
#ifndef USE_ASYNC_LDS
#define USE_ASYNC_LDS 0
#endif

#define AS1 __attribute__((address_space(1)))
#define AS3 __attribute__((address_space(3)))

// Build a 16-half fragment from two aligned 16-byte loads (global or LDS).
__device__ __forceinline__ v16h load_frag(const _Float16* p0, const _Float16* p1) {
  v8h lo = *(const v8h*)p0;
  v8h hi = *(const v8h*)p1;
  v16h r;
#pragma unroll
  for (int i = 0; i < 8; ++i) { r[i] = lo[i]; r[i + 8] = hi[i]; }
  return r;
}

// Branchless top-16 insert: replace the slot equal to curMin (first match)
// with (cand, ci), then recompute curMin with a depth-4 fmin tree.
// Lowers to v_cmp/v_cndmask chains only -- no exec-mask branching per slot.
__device__ __forceinline__ void topk_insert(float (&tv)[16], int (&ti)[16],
                                            float& curMin, float cand, int ci) {
  bool taken = false;
#pragma unroll
  for (int s2 = 0; s2 < 16; ++s2) {
    const bool take = (!taken) && (tv[s2] == curMin);
    tv[s2] = take ? cand : tv[s2];
    ti[s2] = take ? ci : ti[s2];
    taken = taken || take;
  }
  float m[16];
#pragma unroll
  for (int s2 = 0; s2 < 16; ++s2) m[s2] = tv[s2];
#pragma unroll
  for (int st = 8; st >= 1; st >>= 1) {
#pragma unroll
    for (int i = 0; i < 16; ++i)
      if (i < st) m[i] = fminf(m[i], m[i + st]);
  }
  curMin = m[0];
}

// ============ K1: f32 (B,C,N) -> f16 transposed (B,N,C) + squared norms ====
__global__ void cvt_feat_kernel(const float* __restrict__ X,
                                _Float16* __restrict__ XT,
                                float* __restrict__ xx, int N) {
  int n = blockIdx.x * blockDim.x + threadIdx.x;
  int b = blockIdx.y;
  if (n >= N) return;
  const float* xb = X + (size_t)b * C_ * N;
  _Float16* xt = XT + ((size_t)b * N + n) * C_;
  float s = 0.f;
#pragma unroll 4
  for (int c = 0; c < C_; ++c) {
    float v = xb[(size_t)c * N + n];
    s += v * v;
    xt[c] = (_Float16)v;
  }
  xx[(size_t)b * N + n] = s;
}

// ============ K1b: edge-conv weight refactor to f16 ========================
// W (Mout, 2C) -> Wh (2*Mout, C): rows [0,Mout)=W1, [Mout,2Mout)=W2-W1
__global__ void cvt_w_kernel(const float* __restrict__ W,
                             _Float16* __restrict__ Wh, int Mout) {
  int i = blockIdx.x * blockDim.x + threadIdx.x;
  int total = 2 * Mout * C_;
  if (i >= total) return;
  int c = i % C_;
  int o = i / C_;
  float v;
  if (o < Mout) {
    v = W[(size_t)o * (2 * C_) + c];
  } else {
    int oo = o - Mout;
    v = W[(size_t)oo * (2 * C_) + C_ + c] - W[(size_t)oo * (2 * C_) + c];
  }
  Wh[i] = (_Float16)v;
}

// ============ K2: fused Gram-tile (WMMA f16->f32) + top-16 KNN ============
// One wave per 16 query rows. Per column tile: 4 chained WMMAs, epilogue to
// LDS, then a 32-lane top-k scan (each lane owns half a row, cached-min
// branchless insertion). B tiles are double-buffered in LDS via async
// global->LDS loads issued *before* the scan so the DMA overlaps the scan.
__global__ void __launch_bounds__(32)
knn_kernel(const _Float16* __restrict__ XT, const float* __restrict__ xx,
           int* __restrict__ idx, int N) {
  const int lane = threadIdx.x;
  const int b = blockIdx.y;
  const int n0 = blockIdx.x * 16;
  const _Float16* Xb = XT + (size_t)b * N * C_;
  const float* xxb = xx + (size_t)b * N;
  const int Mc     = lane & 15;
  const int hc     = lane >> 4;              // which half-row this lane scans
  const int baseKA = (lane < 16) ? 0 : 8;    // A-matrix lane K split
  const int baseKB = (lane < 16) ? 0 : 8;    // B-matrix lane K split
  const int roff   = (lane < 16) ? 0 : 8;    // C/D row offset

  // A fragments for the 16 query rows, reused across all column tiles
  v16h afrag[4];
  {
    const _Float16* arow = Xb + (size_t)(n0 + Mc) * C_;
#pragma unroll
    for (int kk = 0; kk < 4; ++kk)
      afrag[kk] = load_frag(arow + 32 * kk + baseKA,
                            arow + 32 * kk + baseKA + 16);
  }
  float xrow[8];
#pragma unroll
  for (int r = 0; r < 8; ++r) xrow[r] = xxb[n0 + r + roff];

  // per-lane top-16 over this lane's half-row, with cached running min
  float tv[16]; int ti[16];
#pragma unroll
  for (int s2 = 0; s2 < 16; ++s2) { tv[s2] = -3.0e38f; ti[s2] = 0; }
  float curMin = -3.0e38f;

  __shared__ float dst[16][17];
  __shared__ float mv[32][16];
  __shared__ int   mi[32][16];

#if USE_ASYNC_LDS
  __shared__ __align__(16) _Float16 Bt[2][16 * C_];   // 2 x 4KB double buffer
  // stage column tile into buffer: lane -> 64B slice (4x b128 async)
  const int srow = lane >> 1;
  const int soff = (lane & 1) * 64;                   // halves
  {
    const _Float16* g = Xb + (size_t)srow * C_ + soff;
    _Float16* l = &Bt[0][srow * C_ + soff];
#pragma unroll
    for (int i = 0; i < 4; ++i)
      __builtin_amdgcn_global_load_async_to_lds_b128(
          (AS1 v4i*)(g + 8 * i), (AS3 v4i*)(l + 8 * i), 0, 0);
  }
#endif

  const int ntiles = N >> 4;
  for (int t = 0; t < ntiles; ++t) {
    const int m0 = t << 4;
#if USE_ASYNC_LDS
    const int buf = t & 1;
    if (t + 1 < ntiles) {  // prefetch next tile while we compute + scan
      const _Float16* g = Xb + (size_t)(m0 + 16 + srow) * C_ + soff;
      _Float16* l = &Bt[buf ^ 1][srow * C_ + soff];
#pragma unroll
      for (int i = 0; i < 4; ++i)
        __builtin_amdgcn_global_load_async_to_lds_b128(
            (AS1 v4i*)(g + 8 * i), (AS3 v4i*)(l + 8 * i), 0, 0);
      __builtin_amdgcn_s_wait_asynccnt(4);  // current tile's 4 loads done
    } else {
      __builtin_amdgcn_s_wait_asynccnt(0);
    }
    const _Float16* brow = &Bt[buf][Mc * C_];
#else
    const _Float16* brow = Xb + (size_t)(m0 + Mc) * C_;
#endif
    v8f acc = {};
#pragma unroll
    for (int kk = 0; kk < 4; ++kk) {
      v16h bfrag = load_frag(brow + 32 * kk + baseKB,
                             brow + 32 * kk + baseKB + 16);
      acc = __builtin_amdgcn_wmma_f32_16x16x32_f16(
          false, afrag[kk], false, bfrag, (short)0, acc, false, false);
    }
    const float xcol = xxb[m0 + Mc];
#pragma unroll
    for (int r = 0; r < 8; ++r)
      dst[r + roff][Mc] = 2.0f * acc[r] - xrow[r] - xcol;  // neg squared dist
    __syncthreads();
    // all 32 lanes scan: lane owns row Mc, columns [hc*8, hc*8+8)
#pragma unroll
    for (int j = 0; j < 8; ++j) {
      const float cand = dst[Mc][hc * 8 + j];
      if (cand > curMin)
        topk_insert(tv, ti, curMin, cand, m0 + hc * 8 + j);
    }
    __syncthreads();
  }

  // merge the two half-row lists for each row
#pragma unroll
  for (int s2 = 0; s2 < 16; ++s2) { mv[lane][s2] = tv[s2]; mi[lane][s2] = ti[s2]; }
  __syncthreads();
  if (lane < 16) {
    const float* ov = mv[lane + 16];
    const int*   oi = mi[lane + 16];
#pragma unroll 4
    for (int j = 0; j < 16; ++j) {
      const float cand = ov[j];
      if (cand > curMin)
        topk_insert(tv, ti, curMin, cand, oi[j]);
    }
    int* op = idx + (((size_t)b * N) + n0 + lane) * 16;
#pragma unroll
    for (int s2 = 0; s2 < 16; ++s2) op[s2] = ti[s2];
  }
}

// ============ K3: Y = Wh(Mout2,C) x X(C,N) via WMMA, Y (B,Mout2,N) f32 ====
__global__ void __launch_bounds__(32)
gemm_wx_kernel(const _Float16* __restrict__ Wh, const _Float16* __restrict__ XT,
               float* __restrict__ Y, int Mout2, int N) {
  const int lane = threadIdx.x;
  const int b  = blockIdx.z;
  const int o0 = blockIdx.y * 16;
  const int m0 = blockIdx.x * 16;
  const int Mc     = lane & 15;
  const int baseKA = (lane < 16) ? 0 : 8;
  const int baseKB = (lane < 16) ? 0 : 8;
  const _Float16* Xb   = XT + (size_t)b * N * C_;
  const _Float16* arow = Wh + (size_t)(o0 + Mc) * C_;
  const _Float16* brow = Xb + (size_t)(m0 + Mc) * C_;
  v8f acc = {};
#pragma unroll
  for (int kk = 0; kk < 4; ++kk) {
    v16h afrag = load_frag(arow + 32 * kk + baseKA, arow + 32 * kk + baseKA + 16);
    v16h bfrag = load_frag(brow + 32 * kk + baseKB, brow + 32 * kk + baseKB + 16);
    acc = __builtin_amdgcn_wmma_f32_16x16x32_f16(
        false, afrag, false, bfrag, (short)0, acc, false, false);
  }
  const int roff = (lane < 16) ? 0 : 8;
  float* yb = Y + ((size_t)b * Mout2 + o0) * N + m0 + Mc;
#pragma unroll
  for (int r = 0; r < 8; ++r) yb[(size_t)(r + roff) * N] = acc[r];
}

// ============ K4: scene gather-max: s = relu(b + Y2[n] + max_k Y1[idx]) ====
__global__ void scene_reduce_kernel(const float* __restrict__ Y,
                                    const int* __restrict__ idx,
                                    const float* __restrict__ bs,
                                    float* __restrict__ s) {
  const int o  = threadIdx.x & 31;
  const int nl = threadIdx.x >> 5;          // 8 points per block
  const int n  = blockIdx.x * 8 + nl;
  const int b  = blockIdx.y;
  const float* Yb = Y + (size_t)b * 64 * NS;
  const int* ip = idx + (((size_t)b * NS) + n) * 16;
  const float* y1 = Yb + (size_t)o * NS;
  float mx = -3.0e38f;
#pragma unroll
  for (int k = 0; k < 16; ++k) mx = fmaxf(mx, y1[ip[k]]);
  const float y2 = Yb[(size_t)(32 + o) * NS + n];
  s[((size_t)b * 32 + o) * NS + n] = fmaxf(0.f, bs[o] + y2 + mx);
}

// ============ K5: template gather-max + global max over m -> t (B,128) ====
__global__ void __launch_bounds__(32)
tmpl_reduce_kernel(const float* __restrict__ Y, const int* __restrict__ idx,
                   const float* __restrict__ bt, float* __restrict__ t) {
  const int lane = threadIdx.x;
  const int o = blockIdx.x;                 // 0..127
  const int b = blockIdx.y;
  const float* Yb = Y + (size_t)b * 256 * NT;
  const float* y1 = Yb + (size_t)o * NT;
  const float* y2 = Yb + (size_t)(128 + o) * NT;
  float mx = -3.0e38f;
  for (int m = lane; m < NT; m += 32) {
    const int* ip = idx + (((size_t)b * NT) + m) * 16;
    float m1 = -3.0e38f;
#pragma unroll
    for (int k = 0; k < 16; ++k) m1 = fmaxf(m1, y1[ip[k]]);
    mx = fmaxf(mx, m1 + y2[m]);
  }
#pragma unroll
  for (int off2 = 16; off2 > 0; off2 >>= 1)
    mx = fmaxf(mx, __shfl_xor(mx, off2, 32));
  if (lane == 0) t[(size_t)b * 128 + o] = fmaxf(0.f, bt[o] + mx);
}

// ============ K6: h = relu(t @ Wc1^T + bc1) ================================
__global__ void head1_kernel(const float* __restrict__ t, const float* __restrict__ Wc1,
                             const float* __restrict__ bc1, float* __restrict__ h) {
  int i = blockIdx.x * blockDim.x + threadIdx.x;
  if (i >= B_ * 128) return;
  int b = i / 128, j = i % 128;
  const float* tb = t + (size_t)b * 128;
  const float* w  = Wc1 + (size_t)j * 128;
  float acc = bc1[j];
#pragma unroll 8
  for (int c = 0; c < 128; ++c) acc += tb[c] * w[c];
  h[i] = fmaxf(0.f, acc);
}

// ============ K7: params = h @ Wc2^T + bc2 (B,3201) ========================
__global__ void head2_kernel(const float* __restrict__ h, const float* __restrict__ Wc2,
                             const float* __restrict__ bc2, float* __restrict__ p) {
  int i = blockIdx.x * blockDim.x + threadIdx.x;
  if (i >= B_ * TP) return;
  int b = i / TP, q = i % TP;
  const float* hb = h + (size_t)b * 128;
  const float* w  = Wc2 + (size_t)q * 128;
  float acc = bc2[q];
#pragma unroll 8
  for (int c = 0; c < 128; ++c) acc += hb[c] * w[c];
  p[i] = acc;
}

// ============ K8: per-batch dynamic 32-32-32-32-1 MLP over N ===============
// params layout: W1@0 W2@1024 W3@2048 W4@3072 b1@3104 b2@3136 b3@3168 b4@3200
__global__ void __launch_bounds__(256)
dyn_mlp_kernel(const float* __restrict__ s, const float* __restrict__ params,
               float* __restrict__ out) {
  __shared__ float P[TP];
  const int b = blockIdx.y;
  for (int i = threadIdx.x; i < TP; i += 256) P[i] = params[(size_t)b * TP + i];
  __syncthreads();
  const int n = blockIdx.x * 256 + threadIdx.x;
  const float* sb = s + (size_t)b * 32 * NS + n;
  float xa[32], xb2[32];
#pragma unroll
  for (int i = 0; i < 32; ++i) xa[i] = sb[(size_t)i * NS];
#pragma unroll
  for (int o = 0; o < 32; ++o) {
    float acc = P[3104 + o];
#pragma unroll
    for (int i = 0; i < 32; ++i) acc += P[o * 32 + i] * xa[i];
    xb2[o] = fmaxf(0.f, acc);
  }
#pragma unroll
  for (int o = 0; o < 32; ++o) {
    float acc = P[3136 + o];
#pragma unroll
    for (int i = 0; i < 32; ++i) acc += P[1024 + o * 32 + i] * xb2[i];
    xa[o] = fmaxf(0.f, acc);
  }
#pragma unroll
  for (int o = 0; o < 32; ++o) {
    float acc = P[3168 + o];
#pragma unroll
    for (int i = 0; i < 32; ++i) acc += P[2048 + o * 32 + i] * xa[i];
    xb2[o] = fmaxf(0.f, acc);
  }
  float acc = P[3200];
#pragma unroll
  for (int i = 0; i < 32; ++i) acc += P[3072 + i] * xb2[i];
  out[(size_t)b * NS + n] = acc;
}

// ============================ launcher =====================================
extern "C" void kernel_launch(void* const* d_in, const int* in_sizes, int n_in,
                              void* d_out, int out_size, void* d_ws, size_t ws_size,
                              hipStream_t stream) {
  (void)in_sizes; (void)n_in; (void)out_size; (void)ws_size;
  const float* scene = (const float*)d_in[0];   // (8,128,4096)
  const float* tmpl  = (const float*)d_in[1];   // (8,128,512)
  const float* Ws    = (const float*)d_in[2];   // (32,256)
  const float* bs    = (const float*)d_in[3];   // (32)
  const float* Wt    = (const float*)d_in[4];   // (128,256)
  const float* bt    = (const float*)d_in[5];   // (128)
  const float* Wc1   = (const float*)d_in[6];   // (128,128)
  const float* bc1   = (const float*)d_in[7];   // (128)
  const float* Wc2   = (const float*)d_in[8];   // (3201,128)
  const float* bc2   = (const float*)d_in[9];   // (3201)
  float* out = (float*)d_out;                   // (8,1,4096)

  char* w = (char*)d_ws;
  size_t off = 0;
  auto alloc = [&](size_t bytes) -> void* {
    void* p = (void*)(w + off);
    off += (bytes + 255) & ~(size_t)255;
    return p;
  };
  _Float16* XTs  = (_Float16*)alloc((size_t)B_ * NS * C_ * 2);   // 8 MB
  float*    xxs  = (float*)   alloc((size_t)B_ * NS * 4);
  _Float16* XTt  = (_Float16*)alloc((size_t)B_ * NT * C_ * 2);   // 1 MB
  float*    xxt  = (float*)   alloc((size_t)B_ * NT * 4);
  _Float16* Whs  = (_Float16*)alloc((size_t)64 * C_ * 2);
  _Float16* Wht  = (_Float16*)alloc((size_t)256 * C_ * 2);
  int*      idxs = (int*)     alloc((size_t)B_ * NS * 16 * 4);   // 2 MB
  int*      idxt = (int*)     alloc((size_t)B_ * NT * 16 * 4);
  float*    Ys   = (float*)   alloc((size_t)B_ * 64 * NS * 4);   // 8 MB
  float*    Yt   = (float*)   alloc((size_t)B_ * 256 * NT * 4);  // 4 MB
  float*    sfeat= (float*)   alloc((size_t)B_ * 32 * NS * 4);   // 4 MB
  float*    tfeat= (float*)   alloc((size_t)B_ * 128 * 4);
  float*    hbuf = (float*)   alloc((size_t)B_ * 128 * 4);
  float*    pbuf = (float*)   alloc((size_t)B_ * TP * 4);

  // Stage 1: convert + norms + weight refactor
  cvt_feat_kernel<<<dim3((NS + 255) / 256, B_), 256, 0, stream>>>(scene, XTs, xxs, NS);
  cvt_feat_kernel<<<dim3((NT + 255) / 256, B_), 256, 0, stream>>>(tmpl,  XTt, xxt, NT);
  cvt_w_kernel<<<(2 * 32 * C_ + 255) / 256, 256, 0, stream>>>(Ws, Whs, 32);
  cvt_w_kernel<<<(2 * 128 * C_ + 255) / 256, 256, 0, stream>>>(Wt, Wht, 128);

  // Stage 2: fused WMMA Gram + top-16 (distances never hit HBM)
  knn_kernel<<<dim3(NS / 16, B_), 32, 0, stream>>>(XTs, xxs, idxs, NS);
  knn_kernel<<<dim3(NT / 16, B_), 32, 0, stream>>>(XTt, xxt, idxt, NT);

  // Stage 3: edge conv as two fused GEMMs (W1·X and (W2-W1)·X)
  gemm_wx_kernel<<<dim3(NS / 16,  64 / 16, B_), 32, 0, stream>>>(Whs, XTs, Ys, 64, NS);
  gemm_wx_kernel<<<dim3(NT / 16, 256 / 16, B_), 32, 0, stream>>>(Wht, XTt, Yt, 256, NT);

  // Stage 4: gather + relu + max reductions
  scene_reduce_kernel<<<dim3(NS / 8, B_), 256, 0, stream>>>(Ys, idxs, bs, sfeat);
  tmpl_reduce_kernel<<<dim3(128, B_), 32, 0, stream>>>(Yt, idxt, bt, tfeat);

  // Stage 5: heads + dynamic MLP
  head1_kernel<<<(B_ * 128 + 255) / 256, 256, 0, stream>>>(tfeat, Wc1, bc1, hbuf);
  head2_kernel<<<(B_ * TP + 255) / 256, 256, 0, stream>>>(hbuf, Wc2, bc2, pbuf);
  dyn_mlp_kernel<<<dim3(NS / 256, B_), 256, 0, stream>>>(sfeat, pbuf, out);
}
